// MultiheadAttentionZ_9809705304106
// MI455X (gfx1250) — matrix operand order repaired and checked by reading, not compiled, not run on hardware
//
#include <hip/hip_runtime.h>
#include <hip/hip_bf16.h>

// ---------------------------------------------------------------------------
// MultiheadAttention forward for MI455X (gfx1250), bf16 WMMA pipeline.
// T=2048, B=4, E=1024, H=16, D=64.
//   k1: QKV projection (WMMA bf16, K-slab 64, vectorized staging, prefetch)
//   k2: flash attention fwd (async K staging, parallel online softmax)
//   k3: averaged attention weights (async Q/K staging, WMMA QK^T)
//   k4: out projection (WMMA bf16, K-slab 64)
// ---------------------------------------------------------------------------

typedef __attribute__((ext_vector_type(16))) __bf16 v16bf;
typedef __attribute__((ext_vector_type(8)))  __bf16 v8bf;
typedef __attribute__((ext_vector_type(4)))  __bf16 v4bf;
typedef __attribute__((ext_vector_type(8)))  float  v8f;

#define WMMA_BF16(a, b, c) \
  __builtin_amdgcn_wmma_f32_16x16x32_bf16(false, (a), false, (b), (short)0, (c), false, false)

static constexpr int T_LEN  = 2048;
static constexpr int BSZ_   = 4;
static constexpr int E_     = 1024;
static constexpr int H_     = 16;
static constexpr int D_     = 64;
static constexpr int ROWS   = T_LEN * BSZ_;   // 8192 flattened (t,b) rows
static constexpr int QKV_LD = 3 * E_;         // 3072

__device__ __forceinline__ v4bf cvt4_bf16(float4 x) {
  v4bf r;
  r[0] = (__bf16)x.x; r[1] = (__bf16)x.y; r[2] = (__bf16)x.z; r[3] = (__bf16)x.w;
  return r;
}

// ---- CDNA5 WMMA fragment loaders (16-bit, contiguous k order, same in A and B) ---------

// A-matrix 16x32 (MxK): lane L row = m0+(L&15); half=L>>4 selects K sub-blocks:
//   v0..7 hold K = k0+half*16 .. +15 (contiguous, matches B loader)
__device__ __forceinline__ v16bf lds_load_a16x32(const __bf16* tile, int ld,
                                                 int m0, int k0, int lane) {
  int m    = m0 + (lane & 15);
  int half = (lane >> 4) & 1;
  const __bf16* p = tile + m * ld + k0 + half * 16;
  v8bf lo = *(const v8bf*)(p);
  v8bf hi = *(const v8bf*)(p + 8);
  return __builtin_shufflevector(lo, hi, 0, 1, 2, 3, 4, 5, 6, 7,
                                 8, 9, 10, 11, 12, 13, 14, 15);
}

// B-matrix 32x16 (KxN) where storage is K-contiguous per N ("tile[n][k]"):
// lane L col = n0+(L&15); reads 16 contiguous K elems at k0 + (L>>4)*16.
__device__ __forceinline__ v16bf lds_load_bT16x32(const __bf16* tile, int ld,
                                                  int n0, int k0, int lane) {
  int n    = n0 + (lane & 15);
  int koff = k0 + ((lane >> 4) & 1) * 16;
  return *(const v16bf*)(tile + n * ld + koff);
}

// ---- CDNA5 async global->LDS staging (ASYNCcnt path, ISA 15.18.3 op 98) ----

// Copy a 64x64 bf16 tile (rows strided by row_stride elems in global) into a
// row-major [64][64] LDS tile using GLOBAL_LOAD_ASYNC_TO_LDS_B128.
__device__ __forceinline__ void async_copy_tile64(const __bf16* gbase,
                                                  long row_stride,
                                                  __bf16* lds, int tid) {
#pragma unroll
  for (int c = tid; c < 512; c += 256) {        // 512 x 16B chunks
    int r = c >> 3, cc = (c & 7) * 8;
    const __bf16* g = gbase + (long)r * row_stride + cc;
    unsigned l = (unsigned)(size_t)(lds + r * 64 + cc);
    asm volatile("global_load_async_to_lds_b128 %0, %1, off"
                 :: "v"(l), "v"(g) : "memory");
  }
}

__device__ __forceinline__ void wait_async() {
  asm volatile("s_wait_asynccnt 0x0" ::: "memory");
}

// ---------------------------------------------------------------------------
// Kernel 1: QKV projection.  C[8192,3072] = X[8192,1024] @ W^T + bias,
// q chunk (n<1024) scaled by D^-0.5 = 0.125.  Output bf16.
// Block tile 64x128, K-slab 64; 8 waves each own 32x32 (2x2 WMMA tiles).
// ---------------------------------------------------------------------------
__global__ __launch_bounds__(256) void qkv_proj_kernel(
    const float* __restrict__ X, const float* __restrict__ W,
    const float* __restrict__ bias, __bf16* __restrict__ QKV) {
  __shared__ __attribute__((aligned(32))) __bf16 As[64 * 64];
  __shared__ __attribute__((aligned(32))) __bf16 Bs[128 * 64];

  int tid = threadIdx.x, lane = tid & 31, wave = tid >> 5;
  int m_blk = blockIdx.y * 64, n_blk = blockIdx.x * 128;
  int wm = (wave & 1) * 32, wn = (wave >> 1) * 32;

  v8f acc[2][2] = {};

  for (int k0 = 0; k0 < E_; k0 += 64) {
    __syncthreads();
    for (int i = tid * 4; i < 64 * 64; i += 1024) {       // 4 floats / thread / iter
      int r = i >> 6, c = i & 63;
      *(v4bf*)&As[i] = cvt4_bf16(*(const float4*)&X[(m_blk + r) * E_ + k0 + c]);
    }
    for (int i = tid * 4; i < 128 * 64; i += 1024) {
      int r = i >> 6, c = i & 63;
      *(v4bf*)&Bs[i] = cvt4_bf16(*(const float4*)&W[(long)(n_blk + r) * E_ + k0 + c]);
    }
    __syncthreads();

    if (k0 + 64 < E_) {  // warm L2/WGP$ for the next K-slab while we compute
      if (tid < 64)  __builtin_prefetch(&X[(m_blk + tid) * E_ + k0 + 64], 0, 1);
      if (tid < 128) __builtin_prefetch(&W[(long)(n_blk + tid) * E_ + k0 + 64], 0, 1);
    }

#pragma unroll
    for (int kk = 0; kk < 64; kk += 32) {
      v16bf a0 = lds_load_a16x32(As, 64, wm, kk, lane);
      v16bf a1 = lds_load_a16x32(As, 64, wm + 16, kk, lane);
      v16bf b0 = lds_load_bT16x32(Bs, 64, wn, kk, lane);
      v16bf b1 = lds_load_bT16x32(Bs, 64, wn + 16, kk, lane);
      acc[0][0] = WMMA_BF16(a0, b0, acc[0][0]);
      acc[0][1] = WMMA_BF16(a0, b1, acc[0][1]);
      acc[1][0] = WMMA_BF16(a1, b0, acc[1][0]);
      acc[1][1] = WMMA_BF16(a1, b1, acc[1][1]);
    }
  }

  int half = (lane >> 4) & 1, nl = lane & 15;
#pragma unroll
  for (int i = 0; i < 2; ++i)
#pragma unroll
    for (int j = 0; j < 2; ++j) {
      int n = n_blk + wn + j * 16 + nl;
      float bv = bias[n];
      float scale = (n < E_) ? 0.125f : 1.0f;  // q * head_dim^-0.5
#pragma unroll
      for (int v = 0; v < 8; ++v) {
        int m = m_blk + wm + i * 16 + v + half * 8;
        QKV[(long)m * QKV_LD + n] = (__bf16)((acc[i][j][v] + bv) * scale);
      }
    }
}

// ---------------------------------------------------------------------------
// Kernel 2: flash-style attention per (b, h, 64-row q tile).
// K tile staged via async global->LDS; V staged transposed (16B global loads).
// scores = Q@K^T via WMMA -> LDS, online softmax parallelized 4 threads/row,
// P(bf16)@V via WMMA with running-accumulator rescale.
// ---------------------------------------------------------------------------
__global__ __launch_bounds__(256) void attn_fwd_kernel(
    const __bf16* __restrict__ QKV, __bf16* __restrict__ CTX,
    float* __restrict__ Mws, float* __restrict__ Lws) {
  __shared__ __attribute__((aligned(32))) __bf16 Qs[64 * 64];
  __shared__ __attribute__((aligned(32))) __bf16 Ks[64 * 64];
  __shared__ __attribute__((aligned(32))) __bf16 VsT[64 * 64];  // [d][s]
  __shared__ __attribute__((aligned(32))) __bf16 Ps[64 * 64];
  __shared__ float Ss[64 * 64];
  __shared__ float red[256];
  __shared__ float mrow[64], lrow[64], arow[64];

  int tid = threadIdx.x, lane = tid & 31, wave = tid >> 5;
  int bh = blockIdx.x;
  int b = bh >> 4, h = bh & 15;
  int t0 = blockIdx.y * 64;
  int qcol = h * D_;
  int kcol = E_ + h * D_;
  int vcol = 2 * E_ + h * D_;
  const long row_stride = (long)BSZ_ * QKV_LD;  // elems between s rows

  // stage Q once (async, straight row copy)
  async_copy_tile64(QKV + (long)(t0 * BSZ_ + b) * QKV_LD + qcol, row_stride,
                    Qs, tid);
  if (tid < 64) { mrow[tid] = -1e30f; lrow[tid] = 0.f; }
  wait_async();
  __syncthreads();

  int wm = (wave & 3) * 16;   // 4 m-tiles of 16
  int wn = (wave >> 2) * 32;  // 2 n-groups of 32 (2 tiles each)
  int rr = tid >> 2, q4 = tid & 3;  // softmax: 4 threads per q row
  v8f ctx0 = {}, ctx1 = {};

  for (int st = 0; st < T_LEN / 64; ++st) {
    int s0 = st * 64;
    // K tile: async global->LDS (no VGPR round trip)
    async_copy_tile64(QKV + (long)(s0 * BSZ_ + b) * QKV_LD + kcol, row_stride,
                      Ks, tid);
    // V tile: transpose through VGPRs (16B global loads, b16 LDS scatter)
    for (int i = tid * 8; i < 64 * 64; i += 2048) {
      int r = i >> 6, c = i & 63;
      v8bf v = *(const v8bf*)&QKV[(long)((s0 + r) * BSZ_ + b) * QKV_LD + vcol + c];
#pragma unroll
      for (int j = 0; j < 8; ++j) VsT[(c + j) * 64 + r] = v[j];
    }
    if (st + 1 < T_LEN / 64 && tid < 64) {  // prefetch next s-tile rows
      long grow = (long)((s0 + 64 + tid) * BSZ_ + b) * QKV_LD;
      __builtin_prefetch(&QKV[grow + kcol], 0, 1);
      __builtin_prefetch(&QKV[grow + vcol], 0, 1);
    }
    wait_async();
    __syncthreads();

    // scores = Q @ K^T   (k-dim = D = 64)
    v8f sc0 = {}, sc1 = {};
#pragma unroll
    for (int kk = 0; kk < D_; kk += 32) {
      v16bf a  = lds_load_a16x32(Qs, 64, wm, kk, lane);
      v16bf b0 = lds_load_bT16x32(Ks, 64, wn, kk, lane);
      v16bf b1 = lds_load_bT16x32(Ks, 64, wn + 16, kk, lane);
      sc0 = WMMA_BF16(a, b0, sc0);
      sc1 = WMMA_BF16(a, b1, sc1);
    }
    {
      int half = (lane >> 4) & 1, nl = lane & 15;
#pragma unroll
      for (int v = 0; v < 8; ++v) {
        int r = wm + v + half * 8;
        Ss[r * 64 + wn + nl]      = sc0[v];
        Ss[r * 64 + wn + 16 + nl] = sc1[v];
      }
    }
    __syncthreads();

    // online softmax, 4 threads per row (16 elems each)
    const float* srow = &Ss[rr * 64 + q4 * 16];
    {
      float mx = -1e30f;
#pragma unroll
      for (int c = 0; c < 16; ++c) mx = fmaxf(mx, srow[c]);
      red[tid] = mx;
    }
    __syncthreads();
    if (q4 == 0) {
      float mold = mrow[rr];
      float mnew = fmaxf(fmaxf(fmaxf(red[tid], red[tid + 1]),
                               fmaxf(red[tid + 2], red[tid + 3])), mold);
      mrow[rr] = mnew;
      arow[rr] = __expf(mold - mnew);
    }
    __syncthreads();
    {
      float mnew = mrow[rr];
      float psum = 0.f;
      __bf16* prow = &Ps[rr * 64 + q4 * 16];
#pragma unroll
      for (int c = 0; c < 16; ++c) {
        float p = __expf(srow[c] - mnew);
        psum += p;
        prow[c] = (__bf16)p;
      }
      red[tid] = psum;
    }
    __syncthreads();
    if (q4 == 0) {
      lrow[rr] = lrow[rr] * arow[rr] +
                 ((red[tid] + red[tid + 1]) + (red[tid + 2] + red[tid + 3]));
    }

    // rescale running context, then ctx += P @ V
    {
      int half = (lane >> 4) & 1;
#pragma unroll
      for (int v = 0; v < 8; ++v) {
        float al = arow[wm + v + half * 8];
        ctx0[v] *= al; ctx1[v] *= al;
      }
    }
#pragma unroll
    for (int kk = 0; kk < 64; kk += 32) {
      v16bf a  = lds_load_a16x32(Ps, 64, wm, kk, lane);
      v16bf b0 = lds_load_bT16x32(VsT, 64, wn, kk, lane);
      v16bf b1 = lds_load_bT16x32(VsT, 64, wn + 16, kk, lane);
      ctx0 = WMMA_BF16(a, b0, ctx0);
      ctx1 = WMMA_BF16(a, b1, ctx1);
    }
    __syncthreads();
  }

  {
    int half = (lane >> 4) & 1, nl = lane & 15;
#pragma unroll
    for (int v = 0; v < 8; ++v) {
      int r = wm + v + half * 8;
      float inv = 1.0f / lrow[r];
      long grow = (long)((t0 + r) * BSZ_ + b) * E_;
      CTX[grow + h * D_ + wn + nl]      = (__bf16)(ctx0[v] * inv);
      CTX[grow + h * D_ + wn + 16 + nl] = (__bf16)(ctx1[v] * inv);
    }
  }
  if (tid < 64) {
    int t = t0 + tid;
    Mws[(long)bh * T_LEN + t] = mrow[tid];
    Lws[(long)bh * T_LEN + t] = lrow[tid];
  }
}

// ---------------------------------------------------------------------------
// Kernel 3: avg_weights[b,t,s] = (1/H) * sum_h exp(score - m)/l.
// Per (b, 64x64 t/s tile): recompute scores per head via WMMA, accumulate.
// Q/K tiles staged via async global->LDS; float4 accumulate/writeback.
// ---------------------------------------------------------------------------
__global__ __launch_bounds__(256) void attn_avg_kernel(
    const __bf16* __restrict__ QKV, const float* __restrict__ Mws,
    const float* __restrict__ Lws, float* __restrict__ AVG) {
  __shared__ __attribute__((aligned(32))) __bf16 Qs[64 * 64];
  __shared__ __attribute__((aligned(32))) __bf16 Ks[64 * 64];
  __shared__ float Ss[64 * 64];
  __shared__ float Acc[64 * 64];

  int tid = threadIdx.x, lane = tid & 31, wave = tid >> 5;
  int b = blockIdx.z;
  int t0 = blockIdx.y * 64, s0 = blockIdx.x * 64;
  int wm = (wave & 3) * 16, wn = (wave >> 2) * 32;
  const long row_stride = (long)BSZ_ * QKV_LD;

  for (int i = tid * 4; i < 64 * 64; i += 1024)
    *(float4*)&Acc[i] = make_float4(0.f, 0.f, 0.f, 0.f);

  for (int h = 0; h < H_; ++h) {
    __syncthreads();
    async_copy_tile64(QKV + (long)(t0 * BSZ_ + b) * QKV_LD + h * D_,
                      row_stride, Qs, tid);
    async_copy_tile64(QKV + (long)(s0 * BSZ_ + b) * QKV_LD + E_ + h * D_,
                      row_stride, Ks, tid);
    wait_async();
    __syncthreads();

    v8f sc0 = {}, sc1 = {};
#pragma unroll
    for (int kk = 0; kk < D_; kk += 32) {
      v16bf a  = lds_load_a16x32(Qs, 64, wm, kk, lane);
      v16bf b0 = lds_load_bT16x32(Ks, 64, wn, kk, lane);
      v16bf b1 = lds_load_bT16x32(Ks, 64, wn + 16, kk, lane);
      sc0 = WMMA_BF16(a, b0, sc0);
      sc1 = WMMA_BF16(a, b1, sc1);
    }
    {
      int half = (lane >> 4) & 1, nl = lane & 15;
#pragma unroll
      for (int v = 0; v < 8; ++v) {
        int r = wm + v + half * 8;
        Ss[r * 64 + wn + nl]      = sc0[v];
        Ss[r * 64 + wn + 16 + nl] = sc1[v];
      }
    }
    __syncthreads();

    int bh = b * H_ + h;
    for (int i = tid * 4; i < 64 * 64; i += 1024) {
      int r = i >> 6;
      int t = t0 + r;
      float m = Mws[(long)bh * T_LEN + t];
      float il = 1.0f / Lws[(long)bh * T_LEN + t];
      float4 s = *(const float4*)&Ss[i];
      float4 a = *(float4*)&Acc[i];
      a.x += __expf(s.x - m) * il;
      a.y += __expf(s.y - m) * il;
      a.z += __expf(s.z - m) * il;
      a.w += __expf(s.w - m) * il;
      *(float4*)&Acc[i] = a;
    }
  }
  __syncthreads();

  const float invH = 1.0f / (float)H_;
  for (int i = tid * 4; i < 64 * 64; i += 1024) {
    int r = i >> 6, c = i & 63;
    float4 a = *(float4*)&Acc[i];
    a.x *= invH; a.y *= invH; a.z *= invH; a.w *= invH;
    *(float4*)&AVG[((long)b * T_LEN + (t0 + r)) * T_LEN + (s0 + c)] = a;
  }
}

// ---------------------------------------------------------------------------
// Kernel 4: out projection.  OUT[8192,1024] = CTX(bf16) @ Wout^T + bias (f32).
// Same tiling as kernel 1 (K-slab 64, 8 WMMAs per stage).
// ---------------------------------------------------------------------------
__global__ __launch_bounds__(256) void out_proj_kernel(
    const __bf16* __restrict__ CTX, const float* __restrict__ W,
    const float* __restrict__ bias, float* __restrict__ OUT) {
  __shared__ __attribute__((aligned(32))) __bf16 As[64 * 64];
  __shared__ __attribute__((aligned(32))) __bf16 Bs[128 * 64];

  int tid = threadIdx.x, lane = tid & 31, wave = tid >> 5;
  int m_blk = blockIdx.y * 64, n_blk = blockIdx.x * 128;
  int wm = (wave & 1) * 32, wn = (wave >> 1) * 32;

  v8f acc[2][2] = {};

  for (int k0 = 0; k0 < E_; k0 += 64) {
    __syncthreads();
    for (int i = tid * 8; i < 64 * 64; i += 2048) {   // bf16 source: 16B copies
      int r = i >> 6, c = i & 63;
      *(v8bf*)&As[i] = *(const v8bf*)&CTX[(long)(m_blk + r) * E_ + k0 + c];
    }
    for (int i = tid * 4; i < 128 * 64; i += 1024) {
      int r = i >> 6, c = i & 63;
      *(v4bf*)&Bs[i] = cvt4_bf16(*(const float4*)&W[(long)(n_blk + r) * E_ + k0 + c]);
    }
    __syncthreads();

    if (k0 + 64 < E_) {
      if (tid < 64)  __builtin_prefetch(&CTX[(long)(m_blk + tid) * E_ + k0 + 64], 0, 1);
      if (tid < 128) __builtin_prefetch(&W[(long)(n_blk + tid) * E_ + k0 + 64], 0, 1);
    }

#pragma unroll
    for (int kk = 0; kk < 64; kk += 32) {
      v16bf a0 = lds_load_a16x32(As, 64, wm, kk, lane);
      v16bf a1 = lds_load_a16x32(As, 64, wm + 16, kk, lane);
      v16bf b0 = lds_load_bT16x32(Bs, 64, wn, kk, lane);
      v16bf b1 = lds_load_bT16x32(Bs, 64, wn + 16, kk, lane);
      acc[0][0] = WMMA_BF16(a0, b0, acc[0][0]);
      acc[0][1] = WMMA_BF16(a0, b1, acc[0][1]);
      acc[1][0] = WMMA_BF16(a1, b0, acc[1][0]);
      acc[1][1] = WMMA_BF16(a1, b1, acc[1][1]);
    }
  }

  int half = (lane >> 4) & 1, nl = lane & 15;
#pragma unroll
  for (int i = 0; i < 2; ++i)
#pragma unroll
    for (int j = 0; j < 2; ++j) {
      int n = n_blk + wn + j * 16 + nl;
      float bv = bias[n];
#pragma unroll
      for (int v = 0; v < 8; ++v) {
        int m = m_blk + wm + i * 16 + v + half * 8;
        OUT[(long)m * E_ + n] = acc[i][j][v] + bv;
      }
    }
}

// ---------------------------------------------------------------------------
extern "C" void kernel_launch(void* const* d_in, const int* in_sizes, int n_in,
                              void* d_out, int out_size, void* d_ws, size_t ws_size,
                              hipStream_t stream) {
  const float* query   = (const float*)d_in[0];
  const float* w_in    = (const float*)d_in[3];
  const float* b_in    = (const float*)d_in[4];
  const float* w_out   = (const float*)d_in[5];
  const float* b_out   = (const float*)d_in[6];

  // workspace layout
  char* ws = (char*)d_ws;
  __bf16* qkv = (__bf16*)ws;                       // 8192*3072 bf16 = 48 MiB
  ws += (size_t)ROWS * QKV_LD * sizeof(__bf16);
  __bf16* ctx = (__bf16*)ws;                       // 8192*1024 bf16 = 16 MiB
  ws += (size_t)ROWS * E_ * sizeof(__bf16);
  float* Mws = (float*)ws;                         // 64*2048 f32
  ws += (size_t)BSZ_ * H_ * T_LEN * sizeof(float);
  float* Lws = (float*)ws;

  float* out = (float*)d_out;                        // [T,B,E]
  float* avg = out + (size_t)T_LEN * BSZ_ * E_;      // [B,T,S]

  // 1) QKV projection (q pre-scaled)
  qkv_proj_kernel<<<dim3(QKV_LD / 128, ROWS / 64), 256, 0, stream>>>(
      query, w_in, b_in, qkv);

  // 2) flash attention fwd: context + (m,l) stats
  attn_fwd_kernel<<<dim3(BSZ_ * H_, T_LEN / 64), 256, 0, stream>>>(
      qkv, ctx, Mws, Lws);

  // 3) averaged attention weights
  attn_avg_kernel<<<dim3(T_LEN / 64, T_LEN / 64, BSZ_), 256, 0, stream>>>(
      qkv, Mws, Lws, avg);

  // 4) output projection
  out_proj_kernel<<<dim3(E_ / 128, ROWS / 64), 256, 0, stream>>>(
      ctx, w_out, b_out, out);
}
